// HybridGSCSNN_15831249453648
// MI455X (gfx1250) — 16-bit and 8-bit matrix operand order repaired and checked by reading, compiled, not run on hardware
//
#include <hip/hip_runtime.h>
#include <hip/hip_bf16.h>

typedef __attribute__((ext_vector_type(16))) _Float16 v16h;
typedef __attribute__((ext_vector_type(8)))  float    v8f;
typedef __attribute__((ext_vector_type(8)))  int      v8i;

#define T_STEPS 200
#define N_IN    120
#define N_PROJ  256
#define N_HID   512
#define N_OUT   12
#define BM      16

#define WS_SX   0
#define WS_SW   1024
#define WS_WQ   2048
#define WS_W1   (WS_WQ + 32*1024)
#define WS_WR   (WS_W1 + 256*1024)
#define WS_WO   (WS_WR + 512*1024)

__global__ __launch_bounds__(256) void k_sx(const float* __restrict__ x,
                                            float* __restrict__ sxArr, int B) {
  __shared__ float red[256];
  const int t = blockIdx.x;
  float m = 0.f;
  for (int e = threadIdx.x; e < B * N_IN; e += 256) {
    int b = e / N_IN, i = e - b * N_IN;
    m = fmaxf(m, fabsf(x[((size_t)b * T_STEPS + t) * N_IN + i]));
  }
  red[threadIdx.x] = m;
  __syncthreads();
  for (int s = 128; s > 0; s >>= 1) {
    if ((int)threadIdx.x < s) red[threadIdx.x] = fmaxf(red[threadIdx.x], red[threadIdx.x + s]);
    __syncthreads();
  }
  if (threadIdx.x == 0) sxArr[t] = red[0] / 127.f + 1e-8f;
}

__global__ __launch_bounds__(256) void k_sw(const float* __restrict__ wp,
                                            float* __restrict__ swArr) {
  __shared__ float red[256];
  float m = 0.f;
  for (int e = threadIdx.x; e < N_PROJ * N_IN; e += 256) m = fmaxf(m, fabsf(wp[e]));
  red[threadIdx.x] = m;
  __syncthreads();
  for (int s = 128; s > 0; s >>= 1) {
    if ((int)threadIdx.x < s) red[threadIdx.x] = fmaxf(red[threadIdx.x], red[threadIdx.x + s]);
    __syncthreads();
  }
  if (threadIdx.x == 0) swArr[0] = red[0] / 127.f + 1e-8f;
}

__global__ __launch_bounds__(256) void k_wq(const float* __restrict__ wp,
                                            const float* __restrict__ swArr,
                                            signed char* __restrict__ wq) {
  int idx = blockIdx.x * 256 + threadIdx.x;
  if (idx >= N_PROJ * 128) return;
  int n = idx >> 7, k = idx & 127;
  float inv = 1.f / swArr[0];
  float q = 0.f;
  if (k < N_IN) q = fminf(fmaxf(rintf(wp[n * N_IN + k] * inv), -127.f), 127.f);
  int kt = k >> 6, kk = k & 63;
  int ln = (n & 15) + 16 * ((kk & 31) >> 4);
  int by = (kk & 15) + 16 * (kk >> 5);
  int f  = kt * 16 + (n >> 4);
  wq[((size_t)f * 32 + ln) * 32 + by] = (signed char)(int)q;
}

__global__ __launch_bounds__(256) void k_swzB16(const float* __restrict__ w,
                                                _Float16* __restrict__ dst,
                                                int nRowsValid, int nRowsPad, int nK) {
  int total = nRowsPad * nK;
  for (int idx = blockIdx.x * 256 + threadIdx.x; idx < total; idx += gridDim.x * 256) {
    int h = idx / nK, k = idx - h * nK;
    float v = (h < nRowsValid) ? w[(size_t)h * nK + k] : 0.f;
    int kt = k >> 5, kk = k & 31;
    int nt = h >> 4, n = h & 15;
    int ln = n + 16 * ((kk & 15) >> 3);
    int hf = (kk & 7) + 8 * (kk >> 4);
    int f  = kt * (nRowsPad >> 4) + nt;
    dst[((size_t)f * 32 + ln) * 16 + hf] = (_Float16)v;
  }
}

__global__ __launch_bounds__(256) void snn_main(
    const float* __restrict__ x,
    const float* __restrict__ b_proj,
    const float* __restrict__ alpha,
    const float* __restrict__ rho,
    const float* __restrict__ beta_a,
    const float* __restrict__ beta_o,
    const float* __restrict__ sxArr,
    const float* __restrict__ swArr,
    const signed char* __restrict__ wqF,
    const _Float16* __restrict__ w1F,
    const _Float16* __restrict__ wrF,
    const _Float16* __restrict__ woF,
    float* __restrict__ out) {
  __shared__ __align__(32) signed char xqA[2][32][32];
  __shared__ __align__(32) _Float16    projA[8][32][16];
  __shared__ __align__(32) _Float16    spkA[16][32][16];
  __shared__ float part[8][256];

  const int tid  = threadIdx.x;
  const int wv   = tid >> 5;
  const int lane = tid & 31;
  const int nl   = lane & 15;
  const int gl   = lane >> 4;
  const int b0   = blockIdx.x * BM;

  for (int e = tid; e < 16 * 32 * 16; e += 256) ((_Float16*)spkA)[e] = (_Float16)0.f;

  float alphaL[4], rhoL[4], betaAL[4];
#pragma unroll
  for (int j = 0; j < 4; ++j) {
    int h = (wv * 4 + j) * 16 + nl;
    alphaL[j] = alpha[h]; rhoL[j] = rho[h]; betaAL[j] = beta_a[h];
  }
  float bprojL[2];
#pragma unroll
  for (int j = 0; j < 2; ++j) bprojL[j] = b_proj[(wv * 2 + j) * 16 + nl];
  const float sw = swArr[0];

  v8f v1s[4] = {}, a1s[4] = {}, spkP[4] = {};
  const int nR = tid & 15;
  const float betaO = (nR < N_OUT) ? beta_o[nR] : 0.f;
  float voutR = 0.f, osumR = 0.f;

  for (int t = 0; t < T_STEPS; ++t) {
    const float sx = sxArr[t];
    const float inv_sx = 1.f / sx;

    for (int e = tid; e < BM * 128; e += 256) {
      int m = e >> 7, k = e & 127;
      float v = 0.f;
      if (k < N_IN) {
        const float* px = &x[((size_t)(b0 + m) * T_STEPS + t) * N_IN + k];
        v = *px;
        if (t + 1 < T_STEPS) __builtin_prefetch(px + N_IN, 0, 1);
      }
      float q = fminf(fmaxf(rintf(v * inv_sx), -127.f), 127.f);
      int frag = k >> 6, kk = k & 63;
      xqA[frag][m + 16 * ((kk & 31) >> 4)][(kk & 15) + 16 * (kk >> 5)] = (signed char)(int)q;
    }
    __syncthreads();

    v8i accp[2] = {};
#pragma unroll
    for (int kt = 0; kt < 2; ++kt) {
      v8i aF = *(const v8i*)&xqA[kt][lane][0];
#pragma unroll
      for (int j = 0; j < 2; ++j) {
        int f = kt * 16 + (wv * 2 + j);
        v8i bF = *(const v8i*)(wqF + ((size_t)f * 32 + lane) * 32);
        accp[j] = __builtin_amdgcn_wmma_i32_16x16x64_iu8(true, aF, true, bF, accp[j], false, false);
      }
    }
    const float pscale = sx * sw;
#pragma unroll
    for (int j = 0; j < 2; ++j) {
      int pc  = (wv * 2 + j) * 16 + nl;
      int kt2 = pc >> 5, kk2 = pc & 31;
      int lnB = 16 * ((kk2 & 15) >> 3);
      int hf  = (kk2 & 7) + 8 * (kk2 >> 4);
#pragma unroll
      for (int r = 0; r < 8; ++r) {
        float pv = (float)accp[j][r] * pscale + bprojL[j];
        projA[kt2][(r + 8 * gl) + lnB][hf] = (_Float16)pv;
      }
    }
    __syncthreads();

    v8f acc[4] = {};
#pragma unroll
    for (int kt = 0; kt < 8; ++kt) {
      v16h aF = *(const v16h*)&projA[kt][lane][0];
#pragma unroll
      for (int j = 0; j < 4; ++j) {
        int f = kt * 32 + (wv * 4 + j);
        v16h bF = *(const v16h*)(w1F + ((size_t)f * 32 + lane) * 16);
        acc[j] = __builtin_amdgcn_wmma_f32_16x16x32_f16(false, aF, false, bF, (short)0, acc[j], false, false);
      }
    }
#pragma unroll
    for (int kt = 0; kt < 16; ++kt) {
      v16h aF = *(const v16h*)&spkA[kt][lane][0];
#pragma unroll
      for (int j = 0; j < 4; ++j) {
        int f = kt * 32 + (wv * 4 + j);
        v16h bF = *(const v16h*)(wrF + ((size_t)f * 32 + lane) * 16);
        acc[j] = __builtin_amdgcn_wmma_f32_16x16x32_f16(false, aF, false, bF, (short)0, acc[j], false, false);
      }
    }

#pragma unroll
    for (int j = 0; j < 4; ++j) {
#pragma unroll
      for (int r = 0; r < 8; ++r) {
        float v1n = alphaL[j] * v1s[j][r] * (1.f - spkP[j][r]) + acc[j][r];
        float th  = 1.f + betaAL[j] * a1s[j][r];
        float s   = (v1n - th >= 0.f) ? 1.f : 0.f;
        a1s[j][r] = rhoL[j] * a1s[j][r] + (1.f - rhoL[j]) * s;
        v1s[j][r] = v1n;
        spkP[j][r] = s;
      }
    }
    __syncthreads();

#pragma unroll
    for (int j = 0; j < 4; ++j) {
      int nt = wv * 4 + j;
      int h  = nt * 16 + nl;
      int kt = h >> 5, kk = h & 31;
      int lnB = 16 * ((kk & 15) >> 3);
      int hf  = (kk & 7) + 8 * (kk >> 4);
#pragma unroll
      for (int r = 0; r < 8; ++r)
        spkA[kt][(r + 8 * gl) + lnB][hf] = (_Float16)spkP[j][r];
    }
    __syncthreads();

    v8f accR = {};
#pragma unroll
    for (int q = 0; q < 2; ++q) {
      int kt = wv * 2 + q;
      v16h aF = *(const v16h*)&spkA[kt][lane][0];
      v16h bF = *(const v16h*)(woF + ((size_t)kt * 32 + lane) * 16);
      accR = __builtin_amdgcn_wmma_f32_16x16x32_f16(false, aF, false, bF, (short)0, accR, false, false);
    }
#pragma unroll
    for (int r = 0; r < 8; ++r) part[wv][(r + 8 * gl) * 16 + nl] = accR[r];
    __syncthreads();

    {
      float s = 0.f;
#pragma unroll
      for (int w2 = 0; w2 < 8; ++w2) s += part[w2][tid];
      voutR = betaO * voutR + (1.f - betaO) * s;
      osumR += voutR;
    }
  }

  if (nR < N_OUT)
    out[(size_t)(b0 + (tid >> 4)) * N_OUT + nR] = osumR * (1.f / (float)T_STEPS);
}

extern "C" void kernel_launch(void* const* d_in, const int* in_sizes, int n_in,
                              void* d_out, int out_size, void* d_ws, size_t ws_size,
                              hipStream_t stream) {
  const float* x      = (const float*)d_in[0];
  const float* w_proj = (const float*)d_in[1];
  const float* b_proj = (const float*)d_in[2];
  const float* w1     = (const float*)d_in[3];
  const float* w_rec  = (const float*)d_in[4];
  const float* w_out  = (const float*)d_in[5];
  const float* alpha  = (const float*)d_in[6];
  const float* rho    = (const float*)d_in[7];
  const float* beta_a = (const float*)d_in[8];
  const float* beta_o = (const float*)d_in[9];
  const int B = in_sizes[0] / (T_STEPS * N_IN);

  char* ws = (char*)d_ws;
  float*       sxArr = (float*)(ws + WS_SX);
  float*       swArr = (float*)(ws + WS_SW);
  signed char* wqF   = (signed char*)(ws + WS_WQ);
  _Float16*    w1F   = (_Float16*)(ws + WS_W1);
  _Float16*    wrF   = (_Float16*)(ws + WS_WR);
  _Float16*    woF   = (_Float16*)(ws + WS_WO);

  k_sx<<<T_STEPS, 256, 0, stream>>>(x, sxArr, B);
  k_sw<<<1, 256, 0, stream>>>(w_proj, swArr);
  k_wq<<<(N_PROJ * 128 + 255) / 256, 256, 0, stream>>>(w_proj, swArr, wqF);
  k_swzB16<<<512, 256, 0, stream>>>(w1, w1F, N_HID, N_HID, N_PROJ);
  k_swzB16<<<1024, 256, 0, stream>>>(w_rec, wrF, N_HID, N_HID, N_HID);
  k_swzB16<<<32, 256, 0, stream>>>(w_out, woF, N_OUT, 16, N_HID);

  snn_main<<<B / BM, 256, 0, stream>>>(x, b_proj, alpha, rho, beta_a, beta_o,
                                       sxArr, swArr, wqF, w1F, wrF, woF,
                                       (float*)d_out);
}
